// PrototypeBarlow_12678743458317
// MI455X (gfx1250) — matrix operand order repaired and checked by reading, not compiled, not run on hardware
//
#include <hip/hip_runtime.h>

// Problem constants (match reference)
#define B_SZ    1024
#define D_IMG   12288
#define P_DIM   512
#define N_PROTO 2048
#define LAMBD_V 0.005f
#define EPS_V   1e-5f

typedef __attribute__((ext_vector_type(16))) __bf16 v16bf;
typedef __attribute__((ext_vector_type(8)))  __bf16 v8bf;
typedef __attribute__((ext_vector_type(4)))  __bf16 v4bf;
typedef __attribute__((ext_vector_type(8)))  float  v8f;

// ---------------------------------------------------------------------------
// Elementwise f32 -> bf16 (vectorized x4), grid-stride
// ---------------------------------------------------------------------------
__global__ __launch_bounds__(256)
void pb_convert(const float* __restrict__ in, __bf16* __restrict__ out, long n4) {
  long stride = (long)gridDim.x * blockDim.x;
  for (long i = (long)blockIdx.x * blockDim.x + threadIdx.x; i < n4; i += stride) {
    float4 v = ((const float4*)in)[i];
    v4bf o = { (__bf16)v.x, (__bf16)v.y, (__bf16)v.z, (__bf16)v.w };
    ((v4bf*)out)[i] = o;
  }
}

// ---------------------------------------------------------------------------
// Tiled transpose: f32 [R,C] -> bf16 [C,R]   (R,C multiples of 64)
// ---------------------------------------------------------------------------
__global__ __launch_bounds__(256)
void pb_transpose(const float* __restrict__ in, __bf16* __restrict__ out, int R, int C) {
  __shared__ __bf16 tile[64][72];
  const int C0 = blockIdx.x * 64;
  const int R0 = blockIdx.y * 64;
  const int t  = threadIdx.x;
  const int c  = t & 63;
  const int rb = t >> 6;
#pragma unroll
  for (int i = 0; i < 16; ++i) {
    const int r = rb + i * 4;
    tile[r][c] = (__bf16)in[(size_t)(R0 + r) * C + C0 + c];
  }
  __syncthreads();
  const int r2 = t & 63;
  const int cb = t >> 6;
#pragma unroll
  for (int i = 0; i < 16; ++i) {
    const int c2 = cb + i * 4;
    out[(size_t)(C0 + c2) * R + R0 + r2] = tile[r2][c2];
  }
}

// ---------------------------------------------------------------------------
// Normalize columns ((x-mean)*rstd) of f32 [R,C] and write transposed bf16 [C,R]
// ---------------------------------------------------------------------------
__global__ __launch_bounds__(256)
void pb_normT(const float* __restrict__ in, const float* __restrict__ mean,
              const float* __restrict__ rstd, __bf16* __restrict__ out,
              int R, int C) {
  __shared__ __bf16 tile[64][72];
  const int C0 = blockIdx.x * 64;
  const int R0 = blockIdx.y * 64;
  const int t  = threadIdx.x;
  const int c  = t & 63;
  const int rb = t >> 6;
  const float m  = mean[C0 + c];
  const float rs = rstd[C0 + c];
#pragma unroll
  for (int i = 0; i < 16; ++i) {
    const int r = rb + i * 4;
    const float v = in[(size_t)(R0 + r) * C + C0 + c];
    tile[r][c] = (__bf16)((v - m) * rs);
  }
  __syncthreads();
  const int r2 = t & 63;
  const int cb = t >> 6;
#pragma unroll
  for (int i = 0; i < 16; ++i) {
    const int c2 = cb + i * 4;
    out[(size_t)(C0 + c2) * R + R0 + r2] = tile[r2][c2];
  }
}

// ---------------------------------------------------------------------------
// Row squared-norms of f32 [rows, cols]; one block per row
// ---------------------------------------------------------------------------
__global__ __launch_bounds__(256)
void pb_row_sqnorm(const float* __restrict__ in, float* __restrict__ out, int cols) {
  const int row = blockIdx.x;
  const float* p = in + (size_t)row * cols;
  float s = 0.f;
  for (int c = threadIdx.x; c < cols; c += 256) { float v = p[c]; s += v * v; }
  __shared__ float red[256];
  red[threadIdx.x] = s;
  __syncthreads();
  for (int o = 128; o > 0; o >>= 1) {
    if (threadIdx.x < o) red[threadIdx.x] += red[threadIdx.x + o];
    __syncthreads();
  }
  if (threadIdx.x == 0) out[row] = red[0];
}

// ---------------------------------------------------------------------------
// Per-column mean / 1/(std+eps) of f32 [rows, cols]; one thread per column
// (coalesced: adjacent threads read adjacent addresses each row)
// ---------------------------------------------------------------------------
__global__ __launch_bounds__(256)
void pb_colstats(const float* __restrict__ in, float* __restrict__ mean,
                 float* __restrict__ rstd, int rows, int cols) {
  const int c = blockIdx.x * 256 + threadIdx.x;
  if (c >= cols) return;
  float s = 0.f, sq = 0.f;
  for (int r = 0; r < rows; ++r) {
    float v = in[(size_t)r * cols + c];
    s += v; sq += v * v;
  }
  const float m   = s / (float)rows;
  float var = sq / (float)rows - m * m;
  var = var > 0.f ? var : 0.f;
  mean[c] = m;
  rstd[c] = 1.0f / (sqrtf(var) + EPS_V);
}

// ---------------------------------------------------------------------------
// Column-wise min of (a+b); one thread per column
// ---------------------------------------------------------------------------
__global__ __launch_bounds__(256)
void pb_colmin(const float* __restrict__ a, const float* __restrict__ b,
               float* __restrict__ out, int rows, int cols) {
  const int c = blockIdx.x * 256 + threadIdx.x;
  if (c >= cols) return;
  float mn = 3.402823466e38f;
  for (int r = 0; r < rows; ++r) {
    const size_t idx = (size_t)r * cols + c;
    mn = fminf(mn, a[idx] + b[idx]);
  }
  out[c] = mn;
}

// ---------------------------------------------------------------------------
// Row-wise min of (a+b); one block per row
// ---------------------------------------------------------------------------
__global__ __launch_bounds__(256)
void pb_rowmin(const float* __restrict__ a, const float* __restrict__ b,
               float* __restrict__ out, int cols) {
  const int row = blockIdx.x;
  const float* pa = a + (size_t)row * cols;
  const float* pb = b + (size_t)row * cols;
  float mn = 3.402823466e38f;
  for (int c = threadIdx.x; c < cols; c += 256)
    mn = fminf(mn, pa[c] + pb[c]);
  __shared__ float red[256];
  red[threadIdx.x] = mn;
  __syncthreads();
  for (int o = 128; o > 0; o >>= 1) {
    if (threadIdx.x < o) red[threadIdx.x] = fminf(red[threadIdx.x], red[threadIdx.x + o]);
    __syncthreads();
  }
  if (threadIdx.x == 0) out[row] = red[0];
}

// ---------------------------------------------------------------------------
// Barlow reduction over raw cross-correlation c_raw (scale 1/B applied here):
// accum[1]+=sum c^2 ; accum[2]+=sum_diag (d-1)^2 ; accum[3]+=sum_diag d^2
// ---------------------------------------------------------------------------
__global__ __launch_bounds__(256)
void pb_creduce(const float* __restrict__ c, float* __restrict__ accum, int n) {
  const size_t total = (size_t)n * n;
  const float scale = 1.0f / (float)B_SZ;
  float s1 = 0.f, s2 = 0.f, s3 = 0.f;
  for (size_t i = (size_t)blockIdx.x * 256 + threadIdx.x; i < total;
       i += (size_t)gridDim.x * 256) {
    const float v = c[i] * scale;
    s1 += v * v;
    if (i % (size_t)(n + 1) == 0) {        // diagonal: i = r*(n+1)
      const float d = v - 1.0f;
      s2 += d * d;
      s3 += v * v;
    }
  }
#pragma unroll
  for (int o = 16; o > 0; o >>= 1) {
    s1 += __shfl_xor(s1, o, 32);
    s2 += __shfl_xor(s2, o, 32);
    s3 += __shfl_xor(s3, o, 32);
  }
  if ((threadIdx.x & 31) == 0) {
    atomicAdd(&accum[1], s1);
    atomicAdd(&accum[2], s2);
    atomicAdd(&accum[3], s3);
  }
}

__global__ void pb_zero(float* accum) {
  if (threadIdx.x < 8) accum[threadIdx.x] = 0.f;
}

// ---------------------------------------------------------------------------
// Final combine: scalar loss
// ---------------------------------------------------------------------------
__global__ __launch_bounds__(256)
void pb_final(const float* __restrict__ colmin, const float* __restrict__ rowmin,
              const float* __restrict__ accum, float* __restrict__ out) {
  __shared__ float red[256];
  __shared__ float sA;
  const int t = threadIdx.x;
  float s = 0.f;
  for (int i = t; i < N_PROTO; i += 256) s += colmin[i];
  red[t] = s; __syncthreads();
  for (int o = 128; o > 0; o >>= 1) { if (t < o) red[t] += red[t + o]; __syncthreads(); }
  if (t == 0) sA = red[0];
  __syncthreads();
  s = 0.f;
  for (int i = t; i < B_SZ; i += 256) s += rowmin[i];
  red[t] = s; __syncthreads();
  for (int o = 128; o > 0; o >>= 1) { if (t < o) red[t] += red[t + o]; __syncthreads(); }
  if (t == 0) {
    const float e1    = sA / (float)N_PROTO;
    const float e2    = red[0] / (float)B_SZ;
    const float vae   = accum[0] / (float)B_SZ;
    const float bloss = accum[2] + LAMBD_V * (accum[1] - accum[3]);
    out[0] = e1 + e2 + vae + bloss;
  }
}

// ---------------------------------------------------------------------------
// WMMA bf16 GEMM:  C[M,N](f32) = A[M,K](bf16,row-major) x Bt[N,K](bf16,row-major)^T
// Block tile 128x128, 8 waves, wave tile 64x32 (8 x v_wmma_f32_16x16x32_bf16
// per K-step of 32). MODE: 0=store, 1=store+bf16 copy, 2=sq-dist epilogue,
// 3=fused VAE squared-error accumulate (no store).
// Requires M,N multiples of 128 and K multiple of 32.
// ---------------------------------------------------------------------------
template <int MODE>
__global__ __launch_bounds__(256)
void pb_gemm(const __bf16* __restrict__ A, const __bf16* __restrict__ Bt,
             float* __restrict__ C, __bf16* __restrict__ Cbf,
             const float* __restrict__ auxA, const float* __restrict__ auxB,
             const float* __restrict__ Ref, float* __restrict__ scalarOut,
             int M, int N, int K) {
  __shared__ alignas(64) __bf16 As[128][32];
  __shared__ alignas(64) __bf16 Bs[128][32];

  const int t    = threadIdx.x;
  const int lane = t & 31;
  const int wid  = t >> 5;
  const int m0   = (wid >> 2) * 64;   // wave row tile (2 rows of waves)
  const int n0   = (wid & 3) * 32;    // wave col tile (4 cols of waves)
  const int blockM = blockIdx.y * 128;
  const int blockN = blockIdx.x * 128;
  const int half = lane >> 4;         // 0/1: which half-wave
  const int l15  = lane & 15;
  const int ka   = half * 16;         // A fragment K base (contiguous, matches B)
  const int kb   = half * 16;         // B fragment K base (contiguous, matches A)

  v8f acc[4][2] = {};

  // cooperative tile load: each thread moves 16 bf16 (two b128) per matrix
  const int ldRow = t >> 1;           // 0..127
  const int ldK   = (t & 1) * 16;     // 0 or 16
  const __bf16* Aptr = A  + (size_t)(blockM + ldRow) * K + ldK;
  const __bf16* Bptr = Bt + (size_t)(blockN + ldRow) * K + ldK;

  for (int kk = 0; kk < K; kk += 32) {
    uint4 av0 = *(const uint4*)(Aptr + kk);
    uint4 av1 = *(const uint4*)(Aptr + kk + 8);
    uint4 bv0 = *(const uint4*)(Bptr + kk);
    uint4 bv1 = *(const uint4*)(Bptr + kk + 8);
    if (kk + 32 < K) {                       // CDNA5 global_prefetch_b8
      __builtin_prefetch(Aptr + kk + 32, 0, 1);
      __builtin_prefetch(Bptr + kk + 32, 0, 1);
    }
    __syncthreads();
    *(uint4*)&As[ldRow][ldK]     = av0;
    *(uint4*)&As[ldRow][ldK + 8] = av1;
    *(uint4*)&Bs[ldRow][ldK]     = bv0;
    *(uint4*)&Bs[ldRow][ldK + 8] = bv1;
    __syncthreads();

    v16bf afrag[4];
#pragma unroll
    for (int i = 0; i < 4; ++i) {
      const __bf16* p = &As[m0 + i * 16 + l15][ka];
      v8bf lo = *(const v8bf*)p;          // K = ka .. ka+7
      v8bf hi = *(const v8bf*)(p + 8);    // K = ka+8 .. ka+15
      afrag[i] = __builtin_shufflevector(lo, hi,
                   0,1,2,3,4,5,6,7,8,9,10,11,12,13,14,15);
    }
    v16bf bfrag[2];
#pragma unroll
    for (int j = 0; j < 2; ++j)
      bfrag[j] = *(const v16bf*)&Bs[n0 + j * 16 + l15][kb]; // K = kb..kb+15

#pragma unroll
    for (int i = 0; i < 4; ++i)
#pragma unroll
      for (int j = 0; j < 2; ++j)
        acc[i][j] = __builtin_amdgcn_wmma_f32_16x16x32_bf16(
            false, afrag[i], false, bfrag[j], (short)0, acc[i][j], false, false);
  }

  // epilogue: ISA f32 C/D layout — VGPR r: lanes0-15 M=r, lanes16-31 M=r+8
  float vaeLocal = 0.0f;
#pragma unroll
  for (int i = 0; i < 4; ++i) {
#pragma unroll
    for (int j = 0; j < 2; ++j) {
#pragma unroll
      for (int r = 0; r < 8; ++r) {
        const int row = blockM + m0 + i * 16 + half * 8 + r;
        const int col = blockN + n0 + j * 16 + l15;
        const size_t idx = (size_t)row * N + col;
        const float v = acc[i][j][r];
        if (MODE == 0) {
          C[idx] = v;
        } else if (MODE == 1) {
          C[idx] = v;
          Cbf[idx] = (__bf16)v;
        } else if (MODE == 2) {
          C[idx] = auxA[row] - 2.0f * v + auxB[col];   // ||a||^2 - 2ab + ||b||^2
        } else {
          const float d = v - Ref[idx];
          vaeLocal += d * d;
        }
      }
    }
  }
  if (MODE == 3) {
#pragma unroll
    for (int o = 16; o > 0; o >>= 1) vaeLocal += __shfl_xor(vaeLocal, o, 32);
    if (lane == 0) atomicAdd(scalarOut, vaeLocal);
  }
}

// ---------------------------------------------------------------------------
extern "C" void kernel_launch(void* const* d_in, const int* in_sizes, int n_in,
                              void* d_out, int out_size, void* d_ws, size_t ws_size,
                              hipStream_t stream) {
  (void)in_sizes; (void)n_in; (void)out_size; (void)ws_size;
  const float* ds_one = (const float*)d_in[0];
  const float* ds_two = (const float*)d_in[1];
  const float* W_enc  = (const float*)d_in[2];
  const float* W_dec  = (const float*)d_in[3];
  const float* protos = (const float*)d_in[4];
  float* out = (float*)d_out;

  char* ws = (char*)d_ws;
  size_t off = 0;
  auto alloc = [&](size_t bytes) -> void* {
    void* p = ws + off;
    off += (bytes + 255) & ~(size_t)255;
    return p;
  };

  // Workspace (~93 MB total with aliasing)
  __bf16* dsA_bf  = (__bf16*)alloc((size_t)B_SZ * D_IMG * 2);      // 25.2 MB
  __bf16* dsB_bf  = (__bf16*)alloc((size_t)B_SZ * D_IMG * 2);      // 25.2 MB
  __bf16* WencT   = (__bf16*)alloc((size_t)P_DIM * D_IMG * 2);     // 12.6 MB  [P,D]
  __bf16* WdecT   = (__bf16*)alloc((size_t)D_IMG * P_DIM * 2);     // 12.6 MB  [D,P]
  __bf16* protoBf = (__bf16*)alloc((size_t)N_PROTO * P_DIM * 2);   //  2.1 MB  [NP,P] (== Bt)
  float*  zA      = (float*)alloc((size_t)B_SZ * P_DIM * 4);
  float*  zB      = (float*)alloc((size_t)B_SZ * P_DIM * 4);
  __bf16* zAbf    = (__bf16*)alloc((size_t)B_SZ * P_DIM * 2);
  __bf16* zBbf    = (__bf16*)alloc((size_t)B_SZ * P_DIM * 2);
  float*  protB   = (float*)alloc((size_t)B_SZ * N_PROTO * 4);     //  8.4 MB
  float*  za2     = (float*)alloc(B_SZ * 4);
  float*  zb2     = (float*)alloc(B_SZ * 4);
  float*  p2      = (float*)alloc(N_PROTO * 4);
  float*  meanA   = (float*)alloc(N_PROTO * 4);
  float*  rstdA   = (float*)alloc(N_PROTO * 4);
  float*  meanB   = (float*)alloc(N_PROTO * 4);
  float*  rstdB   = (float*)alloc(N_PROTO * 4);
  float*  colmin  = (float*)alloc(N_PROTO * 4);
  float*  rowmin  = (float*)alloc(B_SZ * 4);
  float*  accum   = (float*)alloc(8 * 4);
  // Aliases with disjoint lifetimes:
  float*  protA = (float*)WencT;                       // written after z GEMMs
  float*  cRaw  = (float*)dsA_bf;                      // written after z_a GEMM (16.8<=25.2 MB)
  __bf16* zanT  = dsB_bf;                              // written after z_b GEMM
  __bf16* zbnT  = dsB_bf + (size_t)N_PROTO * B_SZ;     // 8.4 <= 25.2 MB total

  pb_zero<<<1, 64, 0, stream>>>(accum);

  // precision conversions / layout prep
  const long nds = (long)B_SZ * D_IMG;
  pb_convert<<<4096, 256, 0, stream>>>(ds_one, dsA_bf, nds / 4);
  pb_convert<<<4096, 256, 0, stream>>>(ds_two, dsB_bf, nds / 4);
  pb_convert<<<1024, 256, 0, stream>>>(protos, protoBf, (long)N_PROTO * P_DIM / 4);
  pb_transpose<<<dim3(P_DIM / 64, D_IMG / 64), 256, 0, stream>>>(W_enc, WencT, D_IMG, P_DIM);
  pb_transpose<<<dim3(D_IMG / 64, P_DIM / 64), 256, 0, stream>>>(W_dec, WdecT, P_DIM, D_IMG);
  pb_row_sqnorm<<<N_PROTO, 256, 0, stream>>>(protos, p2, P_DIM);

  // z = ds @ W_enc   [1024,512], keep f32 + bf16 copies
  pb_gemm<1><<<dim3(P_DIM / 128, B_SZ / 128), 256, 0, stream>>>(
      dsA_bf, WencT, zA, zAbf, nullptr, nullptr, nullptr, nullptr, B_SZ, P_DIM, D_IMG);
  pb_gemm<1><<<dim3(P_DIM / 128, B_SZ / 128), 256, 0, stream>>>(
      dsB_bf, WencT, zB, zBbf, nullptr, nullptr, nullptr, nullptr, B_SZ, P_DIM, D_IMG);
  pb_row_sqnorm<<<B_SZ, 256, 0, stream>>>(zA, za2, P_DIM);
  pb_row_sqnorm<<<B_SZ, 256, 0, stream>>>(zB, zb2, P_DIM);

  // prot = ||z||^2 - 2 z@P^T + ||p||^2   [1024,2048] (fused epilogue)
  pb_gemm<2><<<dim3(N_PROTO / 128, B_SZ / 128), 256, 0, stream>>>(
      zAbf, protoBf, protA, nullptr, za2, p2, nullptr, nullptr, B_SZ, N_PROTO, P_DIM);
  pb_gemm<2><<<dim3(N_PROTO / 128, B_SZ / 128), 256, 0, stream>>>(
      zBbf, protoBf, protB, nullptr, zb2, p2, nullptr, nullptr, B_SZ, N_PROTO, P_DIM);

  // prototype loss pieces
  pb_colmin<<<N_PROTO / 256, 256, 0, stream>>>(protA, protB, colmin, B_SZ, N_PROTO);
  pb_rowmin<<<B_SZ, 256, 0, stream>>>(protA, protB, rowmin, N_PROTO);

  // Barlow: column stats, normalize+transpose, c = za_n^T @ zb_n
  pb_colstats<<<N_PROTO / 256, 256, 0, stream>>>(protA, meanA, rstdA, B_SZ, N_PROTO);
  pb_colstats<<<N_PROTO / 256, 256, 0, stream>>>(protB, meanB, rstdB, B_SZ, N_PROTO);
  pb_normT<<<dim3(N_PROTO / 64, B_SZ / 64), 256, 0, stream>>>(protA, meanA, rstdA, zanT, B_SZ, N_PROTO);
  pb_normT<<<dim3(N_PROTO / 64, B_SZ / 64), 256, 0, stream>>>(protB, meanB, rstdB, zbnT, B_SZ, N_PROTO);
  pb_gemm<0><<<dim3(N_PROTO / 128, N_PROTO / 128), 256, 0, stream>>>(
      zanT, zbnT, cRaw, nullptr, nullptr, nullptr, nullptr, nullptr, N_PROTO, N_PROTO, B_SZ);
  pb_creduce<<<4096, 256, 0, stream>>>(cRaw, accum, N_PROTO);

  // VAE: dec = z @ W_dec, fused sum((dec-ds)^2) -> accum[0] (dec never stored)
  pb_gemm<3><<<dim3(D_IMG / 128, B_SZ / 128), 256, 0, stream>>>(
      zAbf, WdecT, nullptr, nullptr, nullptr, nullptr, ds_one, accum, B_SZ, D_IMG, P_DIM);
  pb_gemm<3><<<dim3(D_IMG / 128, B_SZ / 128), 256, 0, stream>>>(
      zBbf, WdecT, nullptr, nullptr, nullptr, nullptr, ds_two, accum, B_SZ, D_IMG, P_DIM);

  pb_final<<<1, 256, 0, stream>>>(colmin, rowmin, accum, out);
}
